// MaskedCausalAttention_70557722738711
// MI455X (gfx1250) — matrix operand order repaired and checked by reading, not compiled, not run on hardware
//
#include <hip/hip_runtime.h>

typedef __attribute__((ext_vector_type(16))) __bf16 v16bf;
typedef __attribute__((ext_vector_type(8)))  float  v8f;

#define TT 2048
#define CC 1024
#define HH 16
#define DD 64
#define BB 4

// Q projection folds (1/sqrt(D)) * log2(e) so softmax runs in exp2 domain.
#define QSCALE 0.180336880f

#if __has_builtin(__builtin_amdgcn_exp2f)
#define EXP2F(x) __builtin_amdgcn_exp2f(x)
#else
#define EXP2F(x) __expf((x)*0.6931471805599453f)
#endif

static __device__ __forceinline__ v8f wmma_bf16(v16bf a, v16bf b, v8f c) {
  return __builtin_amdgcn_wmma_f32_16x16x32_bf16(false, a, false, b, (short)0, c,
                                                 false, false);
}

// ---- DPP16 cross-lane helper: old=0 + bound_ctrl=1 -> no pre-copy, foldable ----
template <int CTRL>
static __device__ __forceinline__ float dppf(float v) {
  int s = __builtin_bit_cast(int, v);
  int d = __builtin_amdgcn_update_dpp(0, s, CTRL, 0xF, 0xF, true);
  return __builtin_bit_cast(float, d);
}
static __device__ __forceinline__ float row16_max(float v) {
  v = fmaxf(v, dppf<0xB1>(v));    // quad_perm [1,0,3,2]  (xor1)
  v = fmaxf(v, dppf<0x4E>(v));    // quad_perm [2,3,0,1]  (xor2)
  v = fmaxf(v, dppf<0x164>(v));   // row_xmask:4
  v = fmaxf(v, dppf<0x168>(v));   // row_xmask:8
  return v;
}

// A fragment (16x32, MxK) from a row-major bf16 tile.
static __device__ __forceinline__ v16bf load_frag_a_lds(const __bf16* tile,
                                                        int rowbase, int stride,
                                                        int kbase, int lane) {
  const int row  = rowbase + (lane & 15);
  const int koff = kbase + ((lane < 16) ? 0 : 16);
  union { v16bf v; uint4 u[2]; } r;
  r.u[0] = *(const uint4*)(tile + row * stride + koff);
  r.u[1] = *(const uint4*)(tile + row * stride + koff + 8);
  return r.v;
}

// B fragment (32x16, KxN) from a tile stored as [n][32 k] bf16.
static __device__ __forceinline__ v16bf load_frag_b_lds(const __bf16* tile,
                                                        int nbase, int lane) {
  const int n    = nbase + (lane & 15);
  const int koff = (lane < 16) ? 0 : 16;
  union { v16bf v; uint4 u[2]; } r;
  r.u[0] = *(const uint4*)(tile + n * 32 + koff);
  r.u[1] = *(const uint4*)(tile + n * 32 + koff + 8);
  return r.v;
}

// B fragment straight from global memory: 32 contiguous bytes per lane.
static __device__ __forceinline__ v16bf load_frag_b_glb(const __bf16* src) {
  union { v16bf v; uint4 u[2]; } r;
  r.u[0] = ((const uint4*)src)[0];
  r.u[1] = ((const uint4*)src)[1];
  return r.v;
}

static __device__ __forceinline__ uint4 pack8(const float* f) {
  union { uint4 u; __bf16 h[8]; } p;
#pragma unroll
  for (int i = 0; i < 8; ++i) p.h[i] = (__bf16)f[i];
  return p.u;
}

// Y[m,n] = sum_k A[m,k] * W[n,k]   (M=8192, N=K=1024), double-buffered LDS.
// MODE 0: -> Q bf16 [B,H,T,D], scaled by QSCALE
// MODE 1: -> K bf16 [B,H,T,D]
// MODE 2: -> V bf16 transposed [B,H,D,T]
// MODE 3: A is bf16 attn, -> f32 out [M,N] with clip(+-10)
template <int MODE>
__global__ __launch_bounds__(256) void gemm_kernel(const void* __restrict__ Ag,
                                                   const float* __restrict__ W,
                                                   void* __restrict__ Yg) {
  __shared__ __bf16 Ash[2][128 * 32];
  __shared__ __bf16 Bsh[2][128 * 32];
  const int lane = threadIdx.x & 31;
  const int wave = threadIdx.x >> 5;
  const int wm   = wave >> 1;   // 0..3 : wave M index
  const int wn   = wave & 1;    // 0..1 : wave N index
  const int m0   = blockIdx.x * 128;
  const int n0   = blockIdx.y * 128;
  const int row  = threadIdx.x >> 1;        // staging: 128 rows, 2 threads/row
  const int colb = (threadIdx.x & 1) * 16;  // 16 contiguous k per thread

  uint4 ar[2], br[2];  // staged tile slices (already bf16-packed)
  auto load_tile = [&](int k0) {
    if (MODE == 3) {
      const __bf16* src = (const __bf16*)Ag + (size_t)(m0 + row) * CC + k0 + colb;
      ar[0] = ((const uint4*)src)[0];
      ar[1] = ((const uint4*)src)[1];
    } else {
      const float* src = (const float*)Ag + (size_t)(m0 + row) * CC + k0 + colb;
      float f[16];
#pragma unroll
      for (int i = 0; i < 16; ++i) f[i] = src[i];
      ar[0] = pack8(f);
      ar[1] = pack8(f + 8);
    }
    const float* ws = W + (size_t)(n0 + row) * CC + k0 + colb;
    float g[16];
#pragma unroll
    for (int i = 0; i < 16; ++i) g[i] = ws[i];
    br[0] = pack8(g);
    br[1] = pack8(g + 8);
  };
  auto store_tile = [&](int buf) {
    *(uint4*)(Ash[buf] + row * 32 + colb)     = ar[0];
    *(uint4*)(Ash[buf] + row * 32 + colb + 8) = ar[1];
    *(uint4*)(Bsh[buf] + row * 32 + colb)     = br[0];
    *(uint4*)(Bsh[buf] + row * 32 + colb + 8) = br[1];
  };

  const v8f vzero = {};
  v8f acc[2][4];
#pragma unroll
  for (int i = 0; i < 2; ++i)
#pragma unroll
    for (int j = 0; j < 4; ++j) acc[i][j] = vzero;

  load_tile(0);
  store_tile(0);
  int cur = 0;
  for (int k0 = 0; k0 < CC; k0 += 32) {
    const bool more = (k0 + 32 < CC);
    if (more) load_tile(k0 + 32);  // global loads stay in flight over compute
    __syncthreads();
    v16bf af[2], bfr[4];
#pragma unroll
    for (int i = 0; i < 2; ++i)
      af[i] = load_frag_a_lds(Ash[cur], wm * 32 + i * 16, 32, 0, lane);
#pragma unroll
    for (int j = 0; j < 4; ++j)
      bfr[j] = load_frag_b_lds(Bsh[cur], wn * 64 + j * 16, lane);
#pragma unroll
    for (int i = 0; i < 2; ++i)
#pragma unroll
      for (int j = 0; j < 4; ++j) acc[i][j] = wmma_bf16(af[i], bfr[j], acc[i][j]);
    if (more) {
      store_tile(cur ^ 1);
      cur ^= 1;
    }
  }

  // ---- epilogue: C layout row = r + 8*(lane>=16), col = lane&15 ----
#pragma unroll
  for (int i = 0; i < 2; ++i)
#pragma unroll
    for (int j = 0; j < 4; ++j) {
      const int nn = n0 + wn * 64 + j * 16 + (lane & 15);
#pragma unroll
      for (int r = 0; r < 8; ++r) {
        const int mm = m0 + wm * 32 + i * 16 + r + ((lane >> 4) << 3);
        float v = acc[i][j][r];
        if (MODE == 0) v *= QSCALE;
        if (MODE == 3) {
          v = fminf(fmaxf(v, -10.0f), 10.0f);
          ((float*)Yg)[(size_t)mm * CC + nn] = v;
        } else {
          const int b = mm >> 11, t = mm & (TT - 1);
          const int h = nn >> 6,  d = nn & (DD - 1);
          size_t idx;
          if (MODE == 2) idx = ((size_t)(b * HH + h) * DD + d) * TT + t;   // V^T
          else           idx = ((size_t)(b * HH + h) * TT + t) * DD + d;   // Q,K
          ((__bf16*)Yg)[idx] = (__bf16)v;
        }
      }
    }
}

// Flash attention: 8 waves/WG, each wave owns 16 query rows; 64-key blocks.
// K fragments pipelined one block ahead; V loads issue before the softmax VALU
// section; row-sums computed on the WMMA pipe via P x ones.
__global__ __launch_bounds__(256) void flash_kernel(const __bf16* __restrict__ Qh,
                                                    const __bf16* __restrict__ Kh,
                                                    const __bf16* __restrict__ Vt,
                                                    __bf16* __restrict__ Attn) {
  __shared__ __bf16 Psh[8 * 16 * 64];  // per-wave 16x64 P tile
  const int lane  = threadIdx.x & 31;
  const int wave  = threadIdx.x >> 5;
  const int b     = blockIdx.z;
  const int h     = blockIdx.y;
  const int qbase = blockIdx.x * 128 + wave * 16;
  const __bf16* Qp = Qh + ((size_t)(b * HH + h) * TT + qbase) * DD;
  const __bf16* Kp = Kh + (size_t)(b * HH + h) * TT * DD;
  const __bf16* Vp = Vt + (size_t)(b * HH + h) * DD * TT;   // [D][T]
  __bf16* pw = Psh + wave * (16 * 64);

  const v8f vzero = {};
  const int c  = lane & 15;
  const int ro = (lane >> 4) << 3;
  const int kqoff = (lane < 16) ? 0 : 16;   // B-frag K-half offset

  // all-ones B fragment for row-sum via WMMA
  v16bf onesb;
#pragma unroll
  for (int i = 0; i < 16; ++i) onesb[i] = (__bf16)1.0f;

  // Q fragments: 16x64 = two 16x32 A fragments (Q carries log2e/sqrt(D))
  v16bf qf[2];
#pragma unroll
  for (int j = 0; j < 2; ++j) {
    const int koff = j * 32 + ((lane < 16) ? 0 : 16);
    union { v16bf v; uint4 u[2]; } r;
    r.u[0] = *(const uint4*)(Qp + c * DD + koff);
    r.u[1] = *(const uint4*)(Qp + c * DD + koff + 8);
    qf[j] = r.v;
  }

  v8f acc[4];
#pragma unroll
  for (int f = 0; f < 4; ++f) acc[f] = vzero;
  float mrow[8], lrow[8], alphav[8];
#pragma unroll
  for (int r = 0; r < 8; ++r) { mrow[r] = -3.0e38f; lrow[r] = 0.0f; }

  v16bf kf[4][2];
  auto load_k = [&](int kb) {
#pragma unroll
    for (int t = 0; t < 4; ++t) {
      const int key = kb + t * 16 + c;
#pragma unroll
      for (int j = 0; j < 2; ++j)
        kf[t][j] = load_frag_b_glb(Kp + (size_t)key * DD + j * 32 + kqoff);
    }
  };

  const int kend = qbase + 16;  // last needed key index is qbase+15
  load_k(0);
  for (int kb = 0; kb < kend; kb += 64) {
    // ---- S = Q * K^T : four 16x16 key tiles, K-dim 64 = 2 WMMA steps each ----
    v8f s[4];
#pragma unroll
    for (int t = 0; t < 4; ++t) {
      s[t] = wmma_bf16(qf[0], kf[t][0], vzero);
      s[t] = wmma_bf16(qf[1], kf[t][1], s[t]);
    }
    // ---- issue V loads for this block, K loads for the next block ----
    v16bf vf[4][2];
#pragma unroll
    for (int f = 0; f < 4; ++f) {
      const int dim = f * 16 + c;
#pragma unroll
      for (int j = 0; j < 2; ++j)
        vf[f][j] = load_frag_b_glb(Vp + (size_t)dim * TT + kb + j * 32 + kqoff);
    }
    if (kb + 64 < kend) load_k(kb + 64);
    // ---- causal mask (only diagonal blocks) ----
    if (kb + 63 > qbase) {
#pragma unroll
      for (int t = 0; t < 4; ++t) {
        const int key = kb + t * 16 + c;
#pragma unroll
        for (int r = 0; r < 8; ++r) {
          if (key > qbase + r + ro) s[t][r] = -3.0e38f;
        }
      }
    }
    // ---- online softmax in exp2 domain (loads in flight above) ----
#pragma unroll
    for (int r = 0; r < 8; ++r) {
      float v = fmaxf(fmaxf(s[0][r], s[1][r]), fmaxf(s[2][r], s[3][r]));
      v = row16_max(v);
      const float mnew = fmaxf(mrow[r], v);
      alphav[r] = EXP2F(mrow[r] - mnew);
      mrow[r] = mnew;
#pragma unroll
      for (int t = 0; t < 4; ++t)
        pw[(r + ro) * 64 + t * 16 + c] = (__bf16)EXP2F(s[t][r] - mnew);
#pragma unroll
      for (int f = 0; f < 4; ++f) acc[f][r] *= alphav[r];
    }
    __builtin_amdgcn_wave_barrier();
    // P (16x64) as two A fragments
    v16bf pfrag[2];
#pragma unroll
    for (int j = 0; j < 2; ++j) pfrag[j] = load_frag_a_lds(pw, 0, 64, j * 32, lane);
    // ---- row sums on the matrix pipe: rsum = P x ones ----
    v8f rsum = wmma_bf16(pfrag[0], onesb, vzero);
    rsum = wmma_bf16(pfrag[1], onesb, rsum);
#pragma unroll
    for (int r = 0; r < 8; ++r) lrow[r] = lrow[r] * alphav[r] + rsum[r];
    // ---- O += P * V ----
#pragma unroll
    for (int f = 0; f < 4; ++f)
#pragma unroll
      for (int j = 0; j < 2; ++j) acc[f] = wmma_bf16(pfrag[j], vf[f][j], acc[f]);
  }

  // ---- normalize (rcp), clip, store attn [B,T,C] bf16 ----
  float linv[8];
#pragma unroll
  for (int r = 0; r < 8; ++r) linv[r] = __builtin_amdgcn_rcpf(lrow[r]);
#pragma unroll
  for (int f = 0; f < 4; ++f) {
    const int d = f * 16 + c;
#pragma unroll
    for (int r = 0; r < 8; ++r) {
      const int qrow = qbase + r + ro;
      float v = acc[f][r] * linv[r];
      v = fminf(fmaxf(v, -10.0f), 10.0f);
      Attn[((size_t)b * TT + qrow) * CC + h * DD + d] = (__bf16)v;
    }
  }
}

extern "C" void kernel_launch(void* const* d_in, const int* in_sizes, int n_in,
                              void* d_out, int out_size, void* d_ws, size_t ws_size,
                              hipStream_t stream) {
  const float* x  = (const float*)d_in[0];
  const float* wq = (const float*)d_in[1];
  const float* wk = (const float*)d_in[3];
  const float* wv = (const float*)d_in[5];
  const float* wp = (const float*)d_in[7];

  const size_t SEG = (size_t)BB * HH * TT * DD;  // 8M bf16 elements = 16 MB
  __bf16* Qh = (__bf16*)d_ws;
  __bf16* Kh = Qh + SEG;
  __bf16* Vt = Kh + SEG;
  __bf16* At = Vt + SEG;

  dim3 g((BB * TT) / 128, CC / 128, 1);  // 64 x 8
  gemm_kernel<0><<<g, 256, 0, stream>>>((const void*)x, wq, (void*)Qh);
  gemm_kernel<1><<<g, 256, 0, stream>>>((const void*)x, wk, (void*)Kh);
  gemm_kernel<2><<<g, 256, 0, stream>>>((const void*)x, wv, (void*)Vt);
  flash_kernel<<<dim3(TT / 128, HH, BB), 256, 0, stream>>>(Qh, Kh, Vt, At);
  gemm_kernel<3><<<g, 256, 0, stream>>>((const void*)At, wp, d_out);
}
